// MambaBlock_28260884808221
// MI455X (gfx1250) — matrix operand order repaired and checked by reading, not compiled, not run on hardware
//
#include <hip/hip_runtime.h>

typedef __attribute__((ext_vector_type(16))) _Float16 v16h;
typedef __attribute__((ext_vector_type(8)))  _Float16 v8h;
typedef __attribute__((ext_vector_type(8)))  float    v8f;

#define B_    2
#define L_    1024
#define DIN_  768
#define E_    1536
#define N_    16
#define R_    48
#define K_    4
#define M_    (B_ * L_)     // 2048 rows for every GEMM
#define TWOE_ (2 * E_)      // 3072
#define SELW_ (2 * N_ + R_) // 80
#define KP_   64            // R_=48 zero-padded to 64 for guard-free WMMA

// ---- WMMA fragment loaders (ISA 7.12.2, 16-bit layouts, wave32) -----------
// A-matrix 16x32 (MxK): lane holds row m=lane&15; lanes0-15 take K=khalf..+7
// in V0..3 and K=khalf+16..+23 in V4..7 (khalf=0), lanes16-31 khalf=8.
// Both halves are 8 contiguous f16 -> two global_load_b128 per fragment.
__device__ inline v16h frag_a(const _Float16* __restrict__ p, int ld, int k0) {
  int lane = threadIdx.x & 31;
  const _Float16* row =
      p + (size_t)(lane & 15) * ld + k0 + ((lane >> 4) << 3);
  v8h lo = *(const v8h*)(row);        // K = base .. base+7
  v8h hi = *(const v8h*)(row + 16);   // K = base+16 .. base+23
  return __builtin_shufflevector(lo, hi, 0, 1, 2, 3, 4, 5, 6, 7,
                                 8, 9, 10, 11, 12, 13, 14, 15);
}

// B-matrix 32x16 (KxN): lane holds column n=lane&15; lanes0-15 K=0..7 and
// K=16..23, lanes16-31 K=8..15 and K=24..31 (same order as A). Fed W stored
// NxK row-major -> two global_load_b128 at +0/+16.
__device__ inline v16h frag_b(const _Float16* __restrict__ p, int ld, int k0) {
  int lane = threadIdx.x & 31;
  const _Float16* row =
      p + (size_t)(lane & 15) * ld + k0 + ((lane >> 4) << 3);
  v8h lo = *(const v8h*)(row);        // K = base .. base+7
  v8h hi = *(const v8h*)(row + 16);   // K = base+16 .. base+23
  return __builtin_shufflevector(lo, hi, 0, 1, 2, 3, 4, 5, 6, 7,
                                 8, 9, 10, 11, 12, 13, 14, 15);
}

// ---- NT-GEMM: C(f32, MxN) = A(f16, MxK) * B(f16, NxK)^T --------------------
// Each wave computes a (16*MT) x (16*NT) tile; 8 waves per 256-thread block
// tile M. A fragments reused across NT, B fragments across MT:
// per K-step 2*MT + 2*NT b128 loads feed MT*NT WMMAs. Kd must be a
// multiple of 32 (operands pre-padded), so no guards -> no divergence.
template <int MT, int NT>
__global__ __launch_bounds__(256) void gemm_nt_wmma(
    const _Float16* __restrict__ A, const _Float16* __restrict__ Bw,
    float* __restrict__ C, int M, int N, int Kd) {
  int wave = threadIdx.x >> 5;
  int lane = threadIdx.x & 31;
  int row0 = (blockIdx.x * 8 + wave) * (16 * MT);
  int col0 = blockIdx.y * (16 * NT);

  v8f acc[MT][NT];
#pragma unroll
  for (int im = 0; im < MT; ++im)
#pragma unroll
    for (int t = 0; t < NT; ++t)
#pragma unroll
      for (int r = 0; r < 8; ++r) acc[im][t][r] = 0.f;

  for (int k0 = 0; k0 < Kd; k0 += 32) {
    v16h a[MT];
#pragma unroll
    for (int im = 0; im < MT; ++im)
      a[im] = frag_a(A + (size_t)(row0 + im * 16) * Kd, Kd, k0);
#pragma unroll
    for (int t = 0; t < NT; ++t) {
      v16h b = frag_b(Bw + (size_t)(col0 + t * 16) * Kd, Kd, k0);
#pragma unroll
      for (int im = 0; im < MT; ++im)
        acc[im][t] = __builtin_amdgcn_wmma_f32_16x16x32_f16(
            false, a[im], false, b, (short)0, acc[im][t], false, false);
    }
  }
  // C/D layout: lane l -> N = l&15, VGPR r -> M = r + (l>=16 ? 8 : 0)
  int rofs = (lane >> 4) << 3;
  int c    = lane & 15;
#pragma unroll
  for (int im = 0; im < MT; ++im)
#pragma unroll
    for (int t = 0; t < NT; ++t)
#pragma unroll
      for (int r = 0; r < 8; ++r)
        C[(size_t)(row0 + im * 16 + rofs + r) * N + col0 + t * 16 + c] =
            acc[im][t][r];
}

// ---- elementwise helpers ---------------------------------------------------
__global__ void cast_f32_to_f16(const float* __restrict__ in,
                                _Float16* __restrict__ out, int n) {
  int i = blockIdx.x * blockDim.x + threadIdx.x;
  if (i < n) out[i] = (_Float16)in[i];
}

// out[r, 0:cols_out] = f16(in[r, col_off : col_off+cols_in]) zero-padded
__global__ void cast_pad_f16(const float* __restrict__ in,
                             _Float16* __restrict__ out, int rows,
                             int cols_in, int col_off, int stride_in,
                             int cols_out) {
  int i = blockIdx.x * blockDim.x + threadIdx.x;
  if (i >= rows * cols_out) return;
  int r = i / cols_out, c = i % cols_out;
  out[i] = (c < cols_in) ? (_Float16)in[(size_t)r * stride_in + col_off + c]
                         : (_Float16)0.f;
}

// causal depthwise conv1d (left pad K-1) + bias + SiLU; emits f32 + f16 copies
__global__ void conv_silu_kernel(const float* __restrict__ xz,
                                 const float* __restrict__ Wc,
                                 const float* __restrict__ bc,
                                 float* __restrict__ xs,
                                 _Float16* __restrict__ xs_h) {
  int i = blockIdx.x * blockDim.x + threadIdx.x;
  if (i >= B_ * L_ * E_) return;
  int e = i % E_;
  int l = (i / E_) % L_;
  int b = i / (E_ * L_);
  float acc = bc[e];
#pragma unroll
  for (int t = 0; t < K_; ++t) {
    int ll = l + t - (K_ - 1);
    if (ll >= 0) acc += xz[(size_t)(b * L_ + ll) * TWOE_ + e] * Wc[e * K_ + t];
  }
  float s = acc / (1.f + __expf(-acc));   // SiLU
  xs[i]   = s;
  xs_h[i] = (_Float16)s;
}

__global__ void softplus_bias_kernel(float* __restrict__ dt,
                                     const float* __restrict__ b_dt) {
  int i = blockIdx.x * blockDim.x + threadIdx.x;
  if (i >= M_ * E_) return;
  float v = dt[i] + b_dt[i % E_];
  dt[i] = (v > 20.f) ? v : log1pf(__expf(v));
}

// selective scan: one thread per (b,e), N=16 state in registers, serial in L.
// Also applies C-dot, D skip, z-gate SiLU; writes f16 for the output GEMM.
__global__ void scan_kernel(const float* __restrict__ dt,
                            const float* __restrict__ xs,
                            const float* __restrict__ sel,
                            const float* __restrict__ A_log,
                            const float* __restrict__ D_param,
                            const float* __restrict__ xz,
                            _Float16* __restrict__ y_h) {
  int i = blockIdx.x * blockDim.x + threadIdx.x;
  if (i >= B_ * E_) return;
  int b = i / E_, e = i % E_;
  float Arow[N_], h[N_];
#pragma unroll
  for (int n = 0; n < N_; ++n) {
    Arow[n] = -__expf(A_log[e * N_ + n]);   // A = -exp(A_log)
    h[n] = 0.f;
  }
  float Dv = D_param[e];
  for (int l = 0; l < L_; ++l) {
    size_t row = (size_t)b * L_ + l;
    float dtv = dt[row * E_ + e];
    float xv  = xs[row * E_ + e];
    float zv  = xz[row * TWOE_ + E_ + e];   // z half of in_proj
    const float* s = sel + row * SELW_;     // [Bm(16) | Cm(16) | dtl(48)]
    float acc = Dv * xv;
    float dx  = dtv * xv;
#pragma unroll
    for (int n = 0; n < N_; ++n) {
      float a = __expf(dtv * Arow[n]);      // ZOH: Ad
      h[n] = a * h[n] + s[n] * dx;          // h = Ad*h + dt*B*x
      acc += h[n] * s[N_ + n];              // y += h . C
    }
    float gz = zv / (1.f + __expf(-zv));    // SiLU(z)
    y_h[row * E_ + e] = (_Float16)(acc * gz);
  }
}

extern "C" void kernel_launch(void* const* d_in, const int* in_sizes, int n_in,
                              void* d_out, int out_size, void* d_ws, size_t ws_size,
                              hipStream_t stream) {
  (void)in_sizes; (void)n_in; (void)out_size; (void)ws_size;
  const float* x      = (const float*)d_in[0];
  const float* W_in   = (const float*)d_in[1];
  const float* W_conv = (const float*)d_in[2];
  const float* b_conv = (const float*)d_in[3];
  const float* W_sel  = (const float*)d_in[4];
  const float* W_dt   = (const float*)d_in[5];
  const float* b_dt   = (const float*)d_in[6];
  const float* A_log  = (const float*)d_in[7];
  const float* D_p    = (const float*)d_in[8];
  const float* W_out  = (const float*)d_in[9];
  float* out = (float*)d_out;

  char* ws = (char*)d_ws;
  size_t off = 0;
  auto take = [&](size_t bytes) -> void* {
    void* p = ws + off;
    off += (bytes + 255) & ~(size_t)255;
    return p;
  };
  _Float16* xh   = (_Float16*)take((size_t)M_ * DIN_ * 2);
  _Float16* Wi_h = (_Float16*)take((size_t)TWOE_ * DIN_ * 2);
  float*    xz   = (float*)   take((size_t)M_ * TWOE_ * 4);
  float*    xs   = (float*)   take((size_t)M_ * E_ * 4);
  _Float16* xs_h = (_Float16*)take((size_t)M_ * E_ * 2);
  _Float16* Ws_h = (_Float16*)take((size_t)SELW_ * E_ * 2);
  float*    sel  = (float*)   take((size_t)M_ * SELW_ * 4);
  _Float16* dtlh = (_Float16*)take((size_t)M_ * KP_ * 2);   // K-padded
  _Float16* Wd_h = (_Float16*)take((size_t)E_ * KP_ * 2);   // K-padded
  float*    dt   = (float*)   take((size_t)M_ * E_ * 4);
  _Float16* y_h  = (_Float16*)take((size_t)M_ * E_ * 2);
  _Float16* Wo_h = (_Float16*)take((size_t)DIN_ * E_ * 2);

  const int TB = 256;
  auto cdiv = [](int a, int b) { return (a + b - 1) / b; };

  // operand casts f32 -> f16 (W_dt padded 48 -> 64 K)
  cast_f32_to_f16<<<cdiv(M_ * DIN_, TB), TB, 0, stream>>>(x, xh, M_ * DIN_);
  cast_f32_to_f16<<<cdiv(TWOE_ * DIN_, TB), TB, 0, stream>>>(W_in, Wi_h, TWOE_ * DIN_);
  cast_f32_to_f16<<<cdiv(SELW_ * E_, TB), TB, 0, stream>>>(W_sel, Ws_h, SELW_ * E_);
  cast_pad_f16<<<cdiv(E_ * KP_, TB), TB, 0, stream>>>(W_dt, Wd_h, E_, R_, 0, R_, KP_);
  cast_f32_to_f16<<<cdiv(DIN_ * E_, TB), TB, 0, stream>>>(W_out, Wo_h, DIN_ * E_);

  // 1) in_proj: xz = x @ W_in.T   (2048 x 3072 x 768)
  gemm_nt_wmma<2, 4><<<dim3(M_ / 256, TWOE_ / 64), 256, 0, stream>>>(
      xh, Wi_h, xz, M_, TWOE_, DIN_);
  // 2) causal depthwise conv + SiLU
  conv_silu_kernel<<<cdiv(B_ * L_ * E_, TB), TB, 0, stream>>>(
      xz, W_conv, b_conv, xs, xs_h);
  // 3) selection: sel = xs @ W_sel.T   (2048 x 80 x 1536); NT=5 covers N=80
  gemm_nt_wmma<2, 5><<<dim3(M_ / 256, 1), 256, 0, stream>>>(
      xs_h, Ws_h, sel, M_, SELW_, E_);
  // dtl = sel[:, 32:80] cast + zero-pad to K=64
  cast_pad_f16<<<cdiv(M_ * KP_, TB), TB, 0, stream>>>(
      sel, dtlh, M_, R_, 2 * N_, SELW_, KP_);
  // 4) dt projection: (2048 x 1536 x 64-padded), fully guard-free
  gemm_nt_wmma<2, 4><<<dim3(M_ / 256, E_ / 64), 256, 0, stream>>>(
      dtlh, Wd_h, dt, M_, E_, KP_);
  softplus_bias_kernel<<<cdiv(M_ * E_, TB), TB, 0, stream>>>(dt, b_dt);
  // 5) selective scan + gating
  scan_kernel<<<cdiv(B_ * E_, TB), TB, 0, stream>>>(
      dt, xs, sel, A_log, D_p, xz, y_h);
  // 6) out_proj: out = y @ W_out.T   (2048 x 768 x 1536)
  gemm_nt_wmma<2, 4><<<dim3(M_ / 256, DIN_ / 64), 256, 0, stream>>>(
      y_h, Wo_h, out, M_, DIN_, E_);
}
